// ExcelFormerConv_1717986919047
// MI455X (gfx1250) — matrix operand order repaired and checked by reading, not compiled, not run on hardware
//
#include <hip/hip_runtime.h>

// Problem constants
#define BATCH   512
#define SEQ     256
#define DIM     256
#define NHEAD   8
#define HDIM    32
#define MROWS   (BATCH * SEQ)          // 131072
#define ATTN_SCALE 0.17677669529663687f // 32^-0.5

typedef __attribute__((ext_vector_type(16))) _Float16 v16h;
typedef __attribute__((ext_vector_type(8)))  float    v8f;

union FragH { v16h v; uint4 q[2]; };
union H8    { _Float16 e[8]; uint4 q; };

// ---- WMMA fragment helpers (gfx1250 16x16x32 f16 layouts) ----
// A (16x32, MxK) row-major source, lda elements:
//   lane<16: M=lane, K = {0..7} in V0-3, {16..23} in V4-7
//   lane>=16: same M, K = {8..15} and {24..31}
__device__ inline v16h frag_a(const _Float16* base, int lda, int lane) {
  int m   = lane & 15;
  int khi = (lane >> 4) << 3;           // 0 or 8
  const _Float16* p = base + (size_t)m * lda + khi;
  FragH f;
  f.q[0] = *(const uint4*)(p);
  f.q[1] = *(const uint4*)(p + 16);
  return f.v;
}
// B (32x16, KxN) supplied as B^T [N,K] row-major, ldb elements:
//   lane<16: N=lane, K = {0..7} and {16..23}; lane>=16: same N, K = {8..15} and {24..31}
__device__ inline v16h frag_b(const _Float16* bt, int ldb, int lane) {
  int n   = lane & 15;
  int klo = (lane >> 4) << 3;           // 0 or 8
  const _Float16* p = bt + (size_t)n * ldb + klo;
  FragH f;
  f.q[0] = *(const uint4*)(p);
  f.q[1] = *(const uint4*)(p + 16);
  return f.v;
}
__device__ inline v8f wmma32(v16h a, v16h b, v8f c) {
  return __builtin_amdgcn_wmma_f32_16x16x32_f16(false, a, false, b, (short)0, c,
                                                false, false);
}

// ---- weight transpose + f32->f16 convert: out[N,K] = (f16) in[K,N] ----
__global__ void transpose_w(const float* __restrict__ in, _Float16* __restrict__ out,
                            int K, int N) {
  int tid = blockIdx.x * blockDim.x + threadIdx.x;
  if (tid >= N * K) return;
  int n = tid / K, k = tid % K;
  out[tid] = (_Float16)in[(size_t)k * N + n];
}

// ---- LayerNorm: one wave per row of 256, f32 in -> f16 out ----
__global__ __launch_bounds__(256) void ln_kernel(
    const float* __restrict__ x, const float* __restrict__ w,
    const float* __restrict__ b, _Float16* __restrict__ h) {
  int wave = threadIdx.x >> 5, lane = threadIdx.x & 31;
  int row = blockIdx.x * 8 + wave;
  const float* xr = x + (size_t)row * DIM;
  int c0 = lane * 8;
  float4 v0 = *(const float4*)(xr + c0);
  float4 v1 = *(const float4*)(xr + c0 + 4);
  float vv[8] = {v0.x, v0.y, v0.z, v0.w, v1.x, v1.y, v1.z, v1.w};
  float s = 0.f, ss = 0.f;
  for (int i = 0; i < 8; ++i) { s += vv[i]; ss += vv[i] * vv[i]; }
  for (int m = 16; m; m >>= 1) {
    s  += __shfl_xor(s, m, 32);
    ss += __shfl_xor(ss, m, 32);
  }
  float mean = s * (1.f / DIM);
  float var  = ss * (1.f / DIM) - mean * mean;
  float rstd = rsqrtf(var + 1e-5f);
  H8 o;
  for (int i = 0; i < 8; ++i)
    o.e[i] = (_Float16)((vv[i] - mean) * rstd * w[c0 + i] + b[c0 + i]);
  *(uint4*)(h + (size_t)row * DIM + c0) = o.q;
}

// ---- QKV GEMM: [MROWS,256]x[256,768] -> scatter to q,k [B,H,S,DH], vT [B,H,DH,S]
__global__ __launch_bounds__(256) void gemm_qkv(
    const _Float16* __restrict__ A, const _Float16* __restrict__ BT,
    _Float16* __restrict__ q, _Float16* __restrict__ k, _Float16* __restrict__ vT) {
  int lane = threadIdx.x & 31, w = threadIdx.x >> 5;
  int wm = w & 3, wn = w >> 2;
  int m0 = blockIdx.y * 128 + wm * 32;
  int n0 = blockIdx.x * 64 + wn * 32;
  v8f acc[2][2] = {};
  const _Float16* ap0 = A + (size_t)m0 * 256;
  const _Float16* ap1 = ap0 + 16 * 256;
  const _Float16* bp0 = BT + (size_t)n0 * 256;
  const _Float16* bp1 = bp0 + 16 * 256;
  for (int k0 = 0; k0 < 256; k0 += 32) {
    v16h a0 = frag_a(ap0 + k0, 256, lane);
    v16h a1 = frag_a(ap1 + k0, 256, lane);
    v16h b0 = frag_b(bp0 + k0, 256, lane);
    v16h b1 = frag_b(bp1 + k0, 256, lane);
    acc[0][0] = wmma32(a0, b0, acc[0][0]);
    acc[0][1] = wmma32(a0, b1, acc[0][1]);
    acc[1][0] = wmma32(a1, b0, acc[1][0]);
    acc[1][1] = wmma32(a1, b1, acc[1][1]);
  }
  int nl = lane & 15, hi = lane >> 4;
  for (int mt = 0; mt < 2; ++mt)
    for (int nt = 0; nt < 2; ++nt) {
      int ncol = n0 + nt * 16 + nl;
      for (int r = 0; r < 8; ++r) {
        int m = m0 + mt * 16 + hi * 8 + r;
        _Float16 val = (_Float16)acc[mt][nt][r];
        int bb = m >> 8, s = m & 255;
        if (ncol < 256) {
          int hh = ncol >> 5, d = ncol & 31;
          q[((((size_t)bb * NHEAD + hh) * SEQ) + s) * HDIM + d] = val;
        } else if (ncol < 512) {
          int n2 = ncol - 256, hh = n2 >> 5, d = n2 & 31;
          k[((((size_t)bb * NHEAD + hh) * SEQ) + s) * HDIM + d] = val;
        } else {
          int n2 = ncol - 512, hh = n2 >> 5, d = n2 & 31;
          vT[((((size_t)bb * NHEAD + hh) * HDIM) + d) * SEQ + s] = val;
        }
      }
    }
}

// ---- Flash attention: 1 WG per (b,h); wave w handles query blocks {w, 15-w} ----
__global__ __launch_bounds__(256) void attn_kernel(
    const _Float16* __restrict__ q, const _Float16* __restrict__ k,
    const _Float16* __restrict__ vT, _Float16* __restrict__ ao) {
  __shared__ _Float16 pstage[8][16][32];  // per-wave prob tile staging (8 KB)
  int lane = threadIdx.x & 31, w = threadIdx.x >> 5;
  int bh = blockIdx.x, b = bh >> 3, h = bh & 7;
  const _Float16* qb = q  + (size_t)bh * SEQ * HDIM;
  const _Float16* kb = k  + (size_t)bh * SEQ * HDIM;
  const _Float16* vb = vT + (size_t)bh * HDIM * SEQ;
  _Float16* aob = ao + ((size_t)b * SEQ) * DIM + h * HDIM;
  int nl = lane & 15, hi = lane >> 4;

  for (int pass = 0; pass < 2; ++pass) {
    int i = pass ? (15 - w) : w;               // query tile index, rows 16i..16i+15
    v16h aq = frag_a(qb + i * 16 * HDIM, HDIM, lane);
    v8f o0 = {}, o1 = {};
    float mrun[8], lrun[8];
    for (int r = 0; r < 8; ++r) { mrun[r] = -1e30f; lrun[r] = 0.f; }

    int nsteps = (i + 2) >> 1;                 // pairs of 16-key tiles, keys <= 16(i+1)
    for (int u = 0; u < nsteps; ++u) {
      int jt0 = 2 * u, jt1 = 2 * u + 1;
      bool have1 = (jt1 <= i);
      v8f c0 = {}, c1 = {};
      v16h bk0 = frag_b(kb + jt0 * 16 * HDIM, HDIM, lane);
      c0 = wmma32(aq, bk0, c0);
      if (have1) {
        v16h bk1 = frag_b(kb + jt1 * 16 * HDIM, HDIM, lane);
        c1 = wmma32(aq, bk1, c1);
      }
      // online softmax over this pair of tiles (half-wave = one row set)
      for (int r = 0; r < 8; ++r) {
        int m = hi * 8 + r, qi = i * 16 + m;
        int kj0 = jt0 * 16 + nl;
        float s0 = c0[r] * ATTN_SCALE + (kj0 > qi ? -10000.f * ATTN_SCALE : 0.f);
        float s1 = -1e30f;
        if (have1) {
          int kj1 = jt1 * 16 + nl;
          s1 = c1[r] * ATTN_SCALE + (kj1 > qi ? -10000.f * ATTN_SCALE : 0.f);
        }
        float tmax = fmaxf(s0, s1);
        for (int d = 8; d; d >>= 1) tmax = fmaxf(tmax, __shfl_xor(tmax, d, 32));
        float nm = fmaxf(mrun[r], tmax);
        float e0 = __expf(s0 - nm);
        float e1 = have1 ? __expf(s1 - nm) : 0.f;
        float rs = e0 + e1;
        for (int d = 8; d; d >>= 1) rs += __shfl_xor(rs, d, 32);
        float corr = __expf(mrun[r] - nm);
        lrun[r] = lrun[r] * corr + rs;
        mrun[r] = nm;
        o0[r] *= corr;
        o1[r] *= corr;
        pstage[w][m][nl]      = (_Float16)e0;
        pstage[w][m][16 + nl] = (_Float16)e1;
      }
      asm volatile("s_wait_dscnt 0" ::: "memory");   // LDS RAW within wave
      v16h ap  = frag_a(&pstage[w][0][0], 32, lane);
      v16h bv0 = frag_b(vb + 32 * u, SEQ, lane);            // dh 0..15
      v16h bv1 = frag_b(vb + 16 * SEQ + 32 * u, SEQ, lane); // dh 16..31
      o0 = wmma32(ap, bv0, o0);
      o1 = wmma32(ap, bv1, o1);
    }
    for (int r = 0; r < 8; ++r) {
      float inv = 1.f / lrun[r];
      int row = i * 16 + hi * 8 + r;
      aob[(size_t)row * DIM + nl]      = (_Float16)(o0[r] * inv);
      aob[(size_t)row * DIM + 16 + nl] = (_Float16)(o1[r] * inv);
    }
  }
}

// ---- Out projection: C = A x woutT + b_out -> f32 (d_out scratch) + f16 copy ----
__global__ __launch_bounds__(256) void gemm_out(
    const _Float16* __restrict__ A, const _Float16* __restrict__ BT,
    const float* __restrict__ bias, float* __restrict__ outf,
    _Float16* __restrict__ outh) {
  int lane = threadIdx.x & 31, w = threadIdx.x >> 5;
  int wm = w & 3, wn = w >> 2;
  int m0 = blockIdx.y * 128 + wm * 32;
  int n0 = blockIdx.x * 64 + wn * 32;
  v8f acc[2][2] = {};
  const _Float16* ap0 = A + (size_t)m0 * 256;
  const _Float16* ap1 = ap0 + 16 * 256;
  const _Float16* bp0 = BT + (size_t)n0 * 256;
  const _Float16* bp1 = bp0 + 16 * 256;
  for (int k0 = 0; k0 < 256; k0 += 32) {
    v16h a0 = frag_a(ap0 + k0, 256, lane);
    v16h a1 = frag_a(ap1 + k0, 256, lane);
    v16h b0 = frag_b(bp0 + k0, 256, lane);
    v16h b1 = frag_b(bp1 + k0, 256, lane);
    acc[0][0] = wmma32(a0, b0, acc[0][0]);
    acc[0][1] = wmma32(a0, b1, acc[0][1]);
    acc[1][0] = wmma32(a1, b0, acc[1][0]);
    acc[1][1] = wmma32(a1, b1, acc[1][1]);
  }
  int nl = lane & 15, hi = lane >> 4;
  for (int mt = 0; mt < 2; ++mt)
    for (int nt = 0; nt < 2; ++nt) {
      int ncol = n0 + nt * 16 + nl;
      float bv = bias[ncol];
      for (int r = 0; r < 8; ++r) {
        int m = m0 + mt * 16 + hi * 8 + r;
        float val = acc[mt][nt][r] + bv;
        size_t idx = (size_t)m * DIM + ncol;
        outf[idx] = val;
        outh[idx] = (_Float16)val;
      }
    }
}

// ---- GLU: g = A x wgluT + b_glu (both halves); d_out += a*tanh(gates) ----
__global__ __launch_bounds__(256) void gemm_glu(
    const _Float16* __restrict__ A, const _Float16* __restrict__ BT,
    const float* __restrict__ bglu, float* __restrict__ io) {
  int lane = threadIdx.x & 31, w = threadIdx.x >> 5;
  int wm = w & 3, wn = w >> 2;
  int m0 = blockIdx.y * 128 + wm * 32;
  int n0 = blockIdx.x * 64 + wn * 32;
  v8f acca[2][2] = {}, accg[2][2] = {};
  const _Float16* ap0 = A + (size_t)m0 * 256;
  const _Float16* ap1 = ap0 + 16 * 256;
  const _Float16* ba0 = BT + (size_t)n0 * 256;          // 'a' weight rows
  const _Float16* ba1 = ba0 + 16 * 256;
  const _Float16* bg0 = BT + (size_t)(256 + n0) * 256;  // gate weight rows
  const _Float16* bg1 = bg0 + 16 * 256;
  for (int k0 = 0; k0 < 256; k0 += 32) {
    v16h a0  = frag_a(ap0 + k0, 256, lane);
    v16h a1  = frag_a(ap1 + k0, 256, lane);
    v16h fa0 = frag_b(ba0 + k0, 256, lane);
    v16h fa1 = frag_b(ba1 + k0, 256, lane);
    v16h fg0 = frag_b(bg0 + k0, 256, lane);
    v16h fg1 = frag_b(bg1 + k0, 256, lane);
    acca[0][0] = wmma32(a0, fa0, acca[0][0]);
    acca[0][1] = wmma32(a0, fa1, acca[0][1]);
    acca[1][0] = wmma32(a1, fa0, acca[1][0]);
    acca[1][1] = wmma32(a1, fa1, acca[1][1]);
    accg[0][0] = wmma32(a0, fg0, accg[0][0]);
    accg[0][1] = wmma32(a0, fg1, accg[0][1]);
    accg[1][0] = wmma32(a1, fg0, accg[1][0]);
    accg[1][1] = wmma32(a1, fg1, accg[1][1]);
  }
  int nl = lane & 15, hi = lane >> 4;
  for (int mt = 0; mt < 2; ++mt)
    for (int nt = 0; nt < 2; ++nt) {
      int ncol = n0 + nt * 16 + nl;
      float bva = bglu[ncol], bvg = bglu[256 + ncol];
      for (int r = 0; r < 8; ++r) {
        int m = m0 + mt * 16 + hi * 8 + r;
        float av = acca[mt][nt][r] + bva;
        float gv = accg[mt][nt][r] + bvg;
        size_t idx = (size_t)m * DIM + ncol;
        io[idx] = io[idx] + av * tanhf(gv);
      }
    }
}

extern "C" void kernel_launch(void* const* d_in, const int* in_sizes, int n_in,
                              void* d_out, int out_size, void* d_ws, size_t ws_size,
                              hipStream_t stream) {
  const float* x     = (const float*)d_in[0];
  const float* ln_w  = (const float*)d_in[1];
  const float* ln_b  = (const float*)d_in[2];
  const float* w_qkv = (const float*)d_in[3];  // [256, 768]
  const float* w_out = (const float*)d_in[4];  // [256, 256]
  const float* b_out = (const float*)d_in[5];
  const float* w_glu = (const float*)d_in[6];  // [256, 512]
  const float* b_glu = (const float*)d_in[7];
  float* out = (float*)d_out;

  char* ws = (char*)d_ws;
  const size_t MB = 1u << 20;
  _Float16* wqkvT = (_Float16*)ws;             // [768,256]
  _Float16* woutT = wqkvT + 768 * 256;         // [256,256]
  _Float16* wgluT = woutT + 256 * 256;         // [512,256]
  _Float16* hbuf  = (_Float16*)(ws + MB);       // h f16 -> reused as out_f16
  _Float16* qbuf  = (_Float16*)(ws + MB + 64 * MB);
  _Float16* kbuf  = (_Float16*)(ws + MB + 128 * MB);
  _Float16* vTbuf = (_Float16*)(ws + MB + 192 * MB);
  _Float16* aobuf = (_Float16*)(ws + MB + 256 * MB);

  transpose_w<<<(768 * 256 + 255) / 256, 256, 0, stream>>>(w_qkv, wqkvT, 256, 768);
  transpose_w<<<(256 * 256 + 255) / 256, 256, 0, stream>>>(w_out, woutT, 256, 256);
  transpose_w<<<(512 * 256 + 255) / 256, 256, 0, stream>>>(w_glu, wgluT, 256, 512);

  ln_kernel<<<MROWS / 8, 256, 0, stream>>>(x, ln_w, ln_b, hbuf);

  gemm_qkv<<<dim3(768 / 64, MROWS / 128), 256, 0, stream>>>(hbuf, wqkvT, qbuf, kbuf, vTbuf);

  attn_kernel<<<BATCH * NHEAD, 256, 0, stream>>>(qbuf, kbuf, vTbuf, aobuf);

  gemm_out<<<dim3(256 / 64, MROWS / 128), 256, 0, stream>>>(aobuf, woutT, b_out, out, hbuf);

  gemm_glu<<<dim3(256 / 64, MROWS / 128), 256, 0, stream>>>(hbuf, wgluT, b_glu, out);
}
